// DSAB_15917148798992
// MI455X (gfx1250) — matrix operand order repaired and checked by reading, not compiled, not run on hardware
//
#include <hip/hip_runtime.h>

typedef __bf16 bf16;
typedef __attribute__((ext_vector_type(16))) __bf16 v16bf;
typedef __attribute__((ext_vector_type(8)))  __bf16 v8bf;
typedef __attribute__((ext_vector_type(2)))  __bf16 v2bf;
typedef __attribute__((ext_vector_type(8)))  float  v8f;

// ---------------------------------------------------------------------------
// WMMA helpers (CDNA5: D = A(16x32 bf16) * B(32x16 bf16) + C(16x16 f32))
// ---------------------------------------------------------------------------
static __device__ __forceinline__ v8f wmma_bf16(v16bf a, v16bf b, v8f c) {
  return __builtin_amdgcn_wmma_f32_16x16x32_bf16(false, a, false, b,
                                                 (short)0, c, false, false);
}

// A-fragment: lane (lr,hi) holds row M=lr, K = kk + hi*16 .. +15 (contiguous)
// p points at element [row][kk + hi*16].
static __device__ __forceinline__ v16bf ldA(const bf16* p) {
  v8bf a = *(const v8bf*)p;
  v8bf b = *(const v8bf*)(p + 8);
  return __builtin_shufflevector(a, b, 0,1,2,3,4,5,6,7,8,9,10,11,12,13,14,15);
}

// B-fragment: lane (lr,hi) holds column N=lr, K = kk + hi*16 .. +15 (contiguous)
static __device__ __forceinline__ v16bf ldB(const bf16* p) {
  return *(const v16bf*)p;
}

// ---------------------------------------------------------------------------
// Kernel 0: zero accumulators (sumsq [4][2][64] f32, attnRaw [4][64][64] f32)
// ---------------------------------------------------------------------------
__global__ void zero_buf(float* __restrict__ sumsq, float* __restrict__ attnRaw) {
  int t = blockIdx.x * 256 + threadIdx.x;
  if (t < 512) sumsq[t] = 0.f;
  for (int i = t; i < 4 * 64 * 64; i += gridDim.x * 256) attnRaw[i] = 0.f;
}

// ---------------------------------------------------------------------------
// Kernel 1: fused qkv = dwconv3x3(conv1x1(x))
//   grid (16, 32, 4)  block 256.  Output tile 16x8, halo 18x10 = 180 positions.
//   conv1x1 as WMMA GEMM (M = halo pos padded to 192, N = 192 oc, K = 64 ic).
//   Ys is [oc][pos] (pos contiguous, stride 194 -> odd bank stride):
//     * WMMA epilogue packs row-pairs -> 4x ds_store_b32 per tile
//     * 3x3 stencil computes a position pair from 6x b32 loads (tap reuse)
//     * Q/K global stores become packed b32 (coalesced pairs of n)
// ---------------------------------------------------------------------------
#define TW 16
#define TH 8
#define HWD 18
#define HHT 10
#define HALO (HHT * HWD) /* 180 */
#define YSTR 194         /* Ys pos stride: 388 B -> odd bank stride */

__global__ __launch_bounds__(256) void qkv_fused(
    const float* __restrict__ x,     // [4,256,256,64]
    const float* __restrict__ wqkv,  // [192,64]
    const float* __restrict__ wdw,   // [192,9]
    bf16* __restrict__ Q, bf16* __restrict__ Kc, bf16* __restrict__ V,
    float* __restrict__ sumsq)       // [4][2][64]
{
  __shared__ bf16  Xs[192 * 64];     // halo inputs (rows 180..191 zero pad)
  __shared__ bf16  Ws[192 * 64];     // qkv 1x1 weights [oc][ic]
  __shared__ bf16  Ys[192 * YSTR];   // conv1x1 result [oc][pos], padded pos
  __shared__ float Wd[192 * 9];      // depthwise weights
  __shared__ float sQ[64], sK[64];

  const int tid = threadIdx.x;
  const int bx = blockIdx.x, by = blockIdx.y, b = blockIdx.z;

  if (tid < 64) { sQ[tid] = 0.f; sK[tid] = 0.f; }
  for (int i = tid; i < (192 - HALO) * 64; i += 256) Xs[HALO * 64 + i] = (bf16)0.f;
  for (int i = tid; i < 192 * 64; i += 256) Ws[i] = (bf16)wqkv[i];
  for (int i = tid; i < 192 * 9;  i += 256) Wd[i] = wdw[i];

  for (int i = tid; i < HALO * 64; i += 256) {
    int hp = i >> 6, c = i & 63;
    int hy = hp / HWD, hx = hp % HWD;
    int gy = by * TH + hy - 1, gx = bx * TW + hx - 1;
    float v = 0.f;
    if ((unsigned)gy < 256u && (unsigned)gx < 256u)
      v = x[(((size_t)b * 256 + gy) * 256 + gx) * 64 + c];
    Xs[i] = (bf16)v;
  }
  __syncthreads();

  // --- GEMM: Ys[192 oc x 192 pos] = (Xs[192 x 64] * Ws^T)^T -----------------
  const int wave = tid >> 5, lane = tid & 31;
  const int lr = lane & 15, hi = lane >> 4;
  for (int job = wave; job < 12 * 12; job += 8) {   // 12 m-tiles x 12 n-tiles
    const int mt = job / 12, nt = job % 12;
    const int m  = mt * 16 + lr;
    const int oc = nt * 16 + lr;
    v8f acc = {};
#pragma unroll
    for (int kk = 0; kk < 64; kk += 32) {
      v16bf a  = ldA(&Xs[m  * 64 + kk + hi * 16]);
      v16bf bb = ldB(&Ws[oc * 64 + kk + hi * 16]);
      acc = wmma_bf16(a, bb, acc);
    }
    const int mbase = mt * 16 + hi * 8;             // even -> b32 aligned
    bf16* ydst = &Ys[oc * YSTR + mbase];
#pragma unroll
    for (int r = 0; r < 8; r += 2) {                // packed pair stores
      v2bf pr; pr[0] = (bf16)acc[r]; pr[1] = (bf16)acc[r + 1];
      *(v2bf*)&ydst[r] = pr;
    }
  }
  __syncthreads();

  // --- depthwise 3x3 on position pairs + scatter to Q/K/V -------------------
  // work item = (oc, pos-pair): 192 * 64 = 12288 items, pos-pair fastest so
  // Q/K packed stores are coalesced across the wave.
  for (int i = tid; i < 192 * 64; i += 256) {
    const int pp = i & 63;
    const int oc = i >> 6;
    const int oy = pp >> 3, ox = (pp & 7) * 2;
    const bf16*  yrow = &Ys[oc * YSTR];
    const float* wr   = &Wd[oc * 9];
    float s0 = 0.f, s1 = 0.f;
#pragma unroll
    for (int dy = 0; dy < 3; ++dy) {
      const int base = (oy + dy) * HWD + ox;        // even -> b32 aligned
      v2bf t0 = *(const v2bf*)&yrow[base];
      v2bf t1 = *(const v2bf*)&yrow[base + 2];
      const float i0 = (float)t0[0], i1 = (float)t0[1];
      const float i2 = (float)t1[0], i3 = (float)t1[1];
      const float w0 = wr[dy * 3], w1 = wr[dy * 3 + 1], w2 = wr[dy * 3 + 2];
      s0 += w0 * i0 + w1 * i1 + w2 * i2;
      s1 += w0 * i1 + w1 * i2 + w2 * i3;
    }
    const int gy = by * TH + oy, gx = bx * TW + ox;
    const size_t n = (size_t)gy * 256 + gx;         // even
    if (oc < 64) {
      v2bf r; r[0] = (bf16)s0; r[1] = (bf16)s1;
      *(v2bf*)&Q[((size_t)b * 64 + oc) * 65536 + n] = r;
      atomicAdd(&sQ[oc], s0 * s0 + s1 * s1);
    } else if (oc < 128) {
      v2bf r; r[0] = (bf16)s0; r[1] = (bf16)s1;
      *(v2bf*)&Kc[((size_t)b * 64 + (oc - 64)) * 65536 + n] = r;
      atomicAdd(&sK[oc - 64], s0 * s0 + s1 * s1);
    } else {
      V[((size_t)b * 65536 + n) * 64 + (oc - 128)]     = (bf16)s0;
      V[((size_t)b * 65536 + n + 1) * 64 + (oc - 128)] = (bf16)s1;
    }
  }
  __syncthreads();
  if (tid < 64) {
    atomicAdd(&sumsq[b * 128 + tid], sQ[tid]);
    atomicAdd(&sumsq[b * 128 + 64 + tid], sK[tid]);
  }
}

// ---------------------------------------------------------------------------
// Kernel 2: Gram matrix attnRaw[b,c,d] += sum_n q[c,n] k[d,n]  (split-K WMMA)
//   grid (32 chunks, 4 batches), block 128 (4 waves); wave = m-tile.
// ---------------------------------------------------------------------------
#define GRAM_CHUNK 2048
__global__ __launch_bounds__(128) void gram(
    const bf16* __restrict__ Q, const bf16* __restrict__ Kc,
    float* __restrict__ attnRaw)
{
  const int b = blockIdx.y;
  const int wave = threadIdx.x >> 5, lane = threadIdx.x & 31;
  const int lr = lane & 15, hi = lane >> 4;
  const int mt = wave;
  const bf16* q = Q  + (size_t)b * 64 * 65536 + (size_t)(mt * 16 + lr) * 65536;
  const bf16* k = Kc + (size_t)b * 64 * 65536;
  v8f a0 = {}, a1 = {}, a2 = {}, a3 = {};
  const int k0 = blockIdx.x * GRAM_CHUNK;
  for (int kk = k0; kk < k0 + GRAM_CHUNK; kk += 32) {
    __builtin_prefetch(&q[kk + 512], 0, 0);          // stream-ahead hint
    v16bf af = ldA(&q[kk + hi * 16]);
    v16bf b0 = ldB(&k[(size_t)(0 * 16 + lr) * 65536 + kk + hi * 16]);
    v16bf b1 = ldB(&k[(size_t)(1 * 16 + lr) * 65536 + kk + hi * 16]);
    v16bf b2 = ldB(&k[(size_t)(2 * 16 + lr) * 65536 + kk + hi * 16]);
    v16bf b3 = ldB(&k[(size_t)(3 * 16 + lr) * 65536 + kk + hi * 16]);
    a0 = wmma_bf16(af, b0, a0);
    a1 = wmma_bf16(af, b1, a1);
    a2 = wmma_bf16(af, b2, a2);
    a3 = wmma_bf16(af, b3, a3);
  }
  float* dst = attnRaw + b * 4096;
#pragma unroll
  for (int r = 0; r < 8; ++r) {
    const int cc = mt * 16 + hi * 8 + r;
    atomicAdd(&dst[cc * 64 +  0 + lr], a0[r]);
    atomicAdd(&dst[cc * 64 + 16 + lr], a1[r]);
    atomicAdd(&dst[cc * 64 + 32 + lr], a2[r]);
    atomicAdd(&dst[cc * 64 + 48 + lr], a3[r]);
  }
}

// ---------------------------------------------------------------------------
// Kernel 3: softmax + fold projection.  grid 4 (per batch), block 256.
//   attn = softmax( raw * rsqrt(sq_q[c]) * rsqrt(sq_k[d]) * T ), M = Wproj@attn
// ---------------------------------------------------------------------------
__global__ __launch_bounds__(256) void softmax_proj(
    const float* __restrict__ attnRaw, const float* __restrict__ sumsq,
    const float* __restrict__ temp, const float* __restrict__ wproj,
    bf16* __restrict__ Mmat)
{
  __shared__ float attnS[64 * 64];
  __shared__ float inq[64], ink[64];
  const int b = blockIdx.x, tid = threadIdx.x;
  if (tid < 64)       inq[tid]      = 1.f / fmaxf(sqrtf(sumsq[b * 128 + tid]), 1e-12f);
  else if (tid < 128) ink[tid - 64] = 1.f / fmaxf(sqrtf(sumsq[b * 128 + tid]), 1e-12f);
  __syncthreads();
  const float T = temp[0];
  if (tid < 64) {
    const int c = tid;
    const float* row = attnRaw + b * 4096 + c * 64;
    const float sc = inq[c] * T;
    float mx = -1e30f;
    for (int d = 0; d < 64; ++d) mx = fmaxf(mx, row[d] * sc * ink[d]);
    float s = 0.f;
    for (int d = 0; d < 64; ++d) s += __expf(row[d] * sc * ink[d] - mx);
    const float inv = 1.f / s;
    for (int d = 0; d < 64; ++d)
      attnS[c * 64 + d] = __expf(row[d] * sc * ink[d] - mx) * inv;
  }
  __syncthreads();
  for (int i = tid; i < 4096; i += 256) {   // M[o][d] = sum_c Wproj[o][c] attn[c][d]
    const int o = i >> 6, d = i & 63;
    float s = 0.f;
    for (int c = 0; c < 64; ++c) s += wproj[o * 64 + c] * attnS[c * 64 + d];
    Mmat[b * 4096 + i] = (bf16)s;
  }
}

// ---------------------------------------------------------------------------
// Kernel 4: out1[o,n] = sum_d M[o,d] v[d,n] -> d_out [b,h,w,c]
//   grid (512, 4), block 256; each wave owns one 16-position tile, all 4 o-tiles.
// ---------------------------------------------------------------------------
__global__ __launch_bounds__(256) void attn_out(
    const bf16* __restrict__ Mmat, const bf16* __restrict__ V,
    float* __restrict__ out)
{
  const int b = blockIdx.y;
  const int wave = threadIdx.x >> 5, lane = threadIdx.x & 31;
  const int lr = lane & 15, hi = lane >> 4;
  const int ntile = blockIdx.x * 8 + wave;
  const size_t n = (size_t)ntile * 16 + lr;
  const bf16* Mb = Mmat + b * 4096;
  const bf16* Vb = V + (size_t)b * 65536 * 64;
  float* ob = out + (size_t)b * 65536 * 64;

  v16bf B0 = ldB(&Vb[n * 64 +  0 + hi * 16]);
  v16bf B1 = ldB(&Vb[n * 64 + 32 + hi * 16]);
#pragma unroll
  for (int mt = 0; mt < 4; ++mt) {
    const int o = mt * 16 + lr;
    v16bf A0 = ldA(&Mb[o * 64 +  0 + hi * 16]);
    v16bf A1 = ldA(&Mb[o * 64 + 32 + hi * 16]);
    v8f acc = {};
    acc = wmma_bf16(A0, B0, acc);
    acc = wmma_bf16(A1, B1, acc);
    float4 lo = make_float4(acc[0], acc[1], acc[2], acc[3]);
    float4 hi4 = make_float4(acc[4], acc[5], acc[6], acc[7]);
    float4* dst = (float4*)&ob[n * 64 + mt * 16 + hi * 8];
    dst[0] = lo;
    dst[1] = hi4;
  }
}

// ---------------------------------------------------------------------------
// Kernels 5/6: positional-encoding path.  Channel-pair per thread (b32 loads),
// unguarded interior fast path (98.5% of positions).
// ---------------------------------------------------------------------------
__global__ __launch_bounds__(256) void pe_conv1(
    const bf16* __restrict__ V, const float* __restrict__ w1,
    bf16* __restrict__ T)
{
  const size_t idx = (size_t)blockIdx.x * 256 + threadIdx.x;  // 4*65536*32
  const int cp = (int)(idx & 31) * 2;
  const size_t pn = idx >> 5;                                  // b*65536 + n
  const int n = (int)(pn & 65535), b = (int)(pn >> 16);
  const int y = n >> 8, xg = n & 255;
  const bf16* Vb = V + (size_t)b * 65536 * 64;
  const float* wc0 = &w1[cp * 9];
  const float* wc1 = &w1[cp * 9 + 9];
  float s0 = 0.f, s1 = 0.f;
  if (y >= 1 && y <= 254 && xg >= 1 && xg <= 254) {
#pragma unroll
    for (int dy = 0; dy < 3; ++dy)
#pragma unroll
      for (int dx = 0; dx < 3; ++dx) {
        v2bf t = *(const v2bf*)&Vb[((size_t)(y + dy - 1) * 256 + (xg + dx - 1)) * 64 + cp];
        s0 += wc0[dy * 3 + dx] * (float)t[0];
        s1 += wc1[dy * 3 + dx] * (float)t[1];
      }
  } else {
#pragma unroll
    for (int dy = 0; dy < 3; ++dy)
#pragma unroll
      for (int dx = 0; dx < 3; ++dx) {
        const int yy = y + dy - 1, xx = xg + dx - 1;
        if ((unsigned)yy < 256u && (unsigned)xx < 256u) {
          v2bf t = *(const v2bf*)&Vb[((size_t)yy * 256 + xx) * 64 + cp];
          s0 += wc0[dy * 3 + dx] * (float)t[0];
          s1 += wc1[dy * 3 + dx] * (float)t[1];
        }
      }
  }
  const float g0 = 0.5f * s0 * (1.f + erff(s0 * 0.70710678118f));
  const float g1 = 0.5f * s1 * (1.f + erff(s1 * 0.70710678118f));
  v2bf r; r[0] = (bf16)g0; r[1] = (bf16)g1;
  *(v2bf*)&T[pn * 64 + cp] = r;
}

__global__ __launch_bounds__(256) void pe_conv2(
    const bf16* __restrict__ T, const float* __restrict__ w2,
    float* __restrict__ out)
{
  const size_t idx = (size_t)blockIdx.x * 256 + threadIdx.x;
  const int cp = (int)(idx & 31) * 2;
  const size_t pn = idx >> 5;
  const int n = (int)(pn & 65535), b = (int)(pn >> 16);
  const int y = n >> 8, xg = n & 255;
  const bf16* Tb = T + (size_t)b * 65536 * 64;
  const float* wc0 = &w2[cp * 9];
  const float* wc1 = &w2[cp * 9 + 9];
  float s0 = 0.f, s1 = 0.f;
  if (y >= 1 && y <= 254 && xg >= 1 && xg <= 254) {
#pragma unroll
    for (int dy = 0; dy < 3; ++dy)
#pragma unroll
      for (int dx = 0; dx < 3; ++dx) {
        v2bf t = *(const v2bf*)&Tb[((size_t)(y + dy - 1) * 256 + (xg + dx - 1)) * 64 + cp];
        s0 += wc0[dy * 3 + dx] * (float)t[0];
        s1 += wc1[dy * 3 + dx] * (float)t[1];
      }
  } else {
#pragma unroll
    for (int dy = 0; dy < 3; ++dy)
#pragma unroll
      for (int dx = 0; dx < 3; ++dx) {
        const int yy = y + dy - 1, xx = xg + dx - 1;
        if ((unsigned)yy < 256u && (unsigned)xx < 256u) {
          v2bf t = *(const v2bf*)&Tb[((size_t)yy * 256 + xx) * 64 + cp];
          s0 += wc0[dy * 3 + dx] * (float)t[0];
          s1 += wc1[dy * 3 + dx] * (float)t[1];
        }
      }
  }
  float2* o = (float2*)&out[pn * 64 + cp];
  float2 v = *o;
  v.x += s0;
  v.y += s1;
  *o = v;
}

// ---------------------------------------------------------------------------
// Launch
// ---------------------------------------------------------------------------
extern "C" void kernel_launch(void* const* d_in, const int* in_sizes, int n_in,
                              void* d_out, int out_size, void* d_ws, size_t ws_size,
                              hipStream_t stream) {
  const float* x     = (const float*)d_in[0];
  const float* wqkv  = (const float*)d_in[1];
  const float* wdw   = (const float*)d_in[2];
  const float* temp  = (const float*)d_in[3];
  const float* wproj = (const float*)d_in[4];
  const float* wpe1  = (const float*)d_in[5];
  const float* wpe2  = (const float*)d_in[6];
  float* out = (float*)d_out;

  const size_t BUF = 33554432ULL;  // 4*64*65536 bf16 = 32 MiB
  char* ws = (char*)d_ws;
  bf16*  Q        = (bf16*)(ws);
  bf16*  Kc       = (bf16*)(ws + BUF);
  bf16*  V        = (bf16*)(ws + 2 * BUF);
  bf16*  T        = (bf16*)(ws + 3 * BUF);
  float* sumsq    = (float*)(ws + 4 * BUF);
  float* attnRaw  = sumsq + 512;
  bf16*  Mmat     = (bf16*)(attnRaw + 4 * 64 * 64);

  zero_buf<<<64, 256, 0, stream>>>(sumsq, attnRaw);
  qkv_fused<<<dim3(16, 32, 4), 256, 0, stream>>>(x, wqkv, wdw, Q, Kc, V, sumsq);
  gram<<<dim3(65536 / GRAM_CHUNK, 4), 128, 0, stream>>>(Q, Kc, attnRaw);
  softmax_proj<<<4, 256, 0, stream>>>(attnRaw, sumsq, temp, wproj, Mmat);
  attn_out<<<dim3(512, 4), 256, 0, stream>>>(Mmat, V, out);
  pe_conv1<<<32768, 256, 0, stream>>>(V, wpe1, T);
  pe_conv2<<<32768, 256, 0, stream>>>(T, wpe2, out);
}
